// CompressedAttention_43069932044593
// MI455X (gfx1250) — matrix operand order repaired and checked by reading, not compiled, not run on hardware
//
#include <hip/hip_runtime.h>

// ---------------------------------------------------------------------------
// CompressedAttention for gfx1250 (CDNA5, wave32, WMMA bf16 16x16x32)
// Rolled, explicitly ping-pong-pipelined WMMA loops: fragment buffers are
// loop-carried so the register allocator cannot collapse the double buffer.
// ---------------------------------------------------------------------------

typedef unsigned short u16;
typedef unsigned int u32;
typedef __attribute__((ext_vector_type(16))) __bf16 v16bf;
typedef __attribute__((ext_vector_type(8)))  float  v8f;
typedef __attribute__((ext_vector_type(4)))  u32    u32x4;

#define D_MODEL   1024
#define N_HEAD    16
#define HD        64
#define NB        2
#define SEQ       2048
#define ROWS      (NB*SEQ)          // 4096 tokens

#define ATTN_PAD  4194304           // 2^22
#define ATTN_KEEP 524288
#define ATTN_N    (D_MODEL*3*D_MODEL)
#define PROJ_PAD  1048576           // 2^20
#define PROJ_KEEP 131072
#define PROJ_N    (D_MODEL*D_MODEL)

__device__ __forceinline__ u16 f2bf(float f) {
  u32 u = __float_as_uint(f);
  u = (u + 0x7fffu + ((u >> 16) & 1u)) >> 16;
  return (u16)u;
}

union Frag {
  v16bf v;
  u16   u[16];
  u32x4 q[2];
};

__device__ __forceinline__ v8f wmma_bf16(v16bf a, v16bf b, v8f c) {
  // (neg_a, A, neg_b, B, c_mod, C, reuse_a, reuse_b)
  return __builtin_amdgcn_wmma_f32_16x16x32_bf16(false, a, false, b, (short)0, c,
                                                 false, false);
}

// A-frag (16xK row-major source): lane = m + 16g;
// halves[0..15] = row[16g .. 16g+15]
__device__ __forceinline__ v16bf load_afrag(const u16* __restrict__ row, int g) {
  Frag f;
  f.q[0] = *(const u32x4*)(row + 16 * g);
  f.q[1] = *(const u32x4*)(row + 16 * g + 8);
  return f.v;
}

// B-frag (KxN, source transposed/row-per-column): lane = n + 16g;
// halves[0..15] = colrow[16g .. 16g+15]
__device__ __forceinline__ v16bf load_bfrag(const u16* __restrict__ colrow, int g) {
  Frag f;
  f.q[0] = *(const u32x4*)(colrow + 16 * g);
  f.q[1] = *(const u32x4*)(colrow + 16 * g + 8);
  return f.v;
}

// ---------------------------------------------------------------------------
// FFT-based weight decompression
// ---------------------------------------------------------------------------

__global__ void zero_c(float2* __restrict__ p, int n) {
  int i = blockIdx.x * blockDim.x + threadIdx.x;
  if (i < n) p[i] = make_float2(0.f, 0.f);
}

// Scatter kept half-spectrum coeffs + Hermitian mirror into full length-n buffer.
__global__ void scatter_herm(const float* __restrict__ re, const float* __restrict__ im,
                             const int* __restrict__ idx, float2* __restrict__ F,
                             int keep, int n) {
  int i = blockIdx.x * blockDim.x + threadIdx.x;
  if (i >= keep) return;
  int k = idx[i];
  float2 c = make_float2(re[i], im[i]);
  F[k] = c;
  if (k > 0 && k < (n >> 1)) F[n - k] = make_float2(c.x, -c.y);
}

__device__ __forceinline__ float2 cmul_tw(float2 v, float c, float s) {
  return make_float2(v.x * c - v.y * s, v.x * s + v.y * c);
}

// One Stockham radix-4 autosort pass of an *inverse* (sign=+1) C2C FFT.
__global__ void ifft_r4(const float2* __restrict__ src, float2* __restrict__ dst,
                        int n, int lg) {
  int m = n >> 2;
  int j = blockIdx.x * blockDim.x + threadIdx.x;
  if (j >= m) return;
  int ns = 1 << lg;
  int t  = j & (ns - 1);

  float2 v0 = src[j];
  float2 v1 = src[j + m];
  float2 v2 = src[j + 2 * m];
  float2 v3 = src[j + 3 * m];

  double a0 = 6.283185307179586476925286766559 * (double)t / (double)(4 * ns);
  float s1, c1, s2, c2, s3, c3;
  __sincosf((float)a0, &s1, &c1);
  __sincosf((float)(2.0 * a0), &s2, &c2);
  __sincosf((float)(3.0 * a0), &s3, &c3);
  v1 = cmul_tw(v1, c1, s1);
  v2 = cmul_tw(v2, c2, s2);
  v3 = cmul_tw(v3, c3, s3);

  float2 t0 = make_float2(v0.x + v2.x, v0.y + v2.y);
  float2 t1 = make_float2(v0.x - v2.x, v0.y - v2.y);
  float2 t2 = make_float2(v1.x + v3.x, v1.y + v3.y);
  float2 t3 = make_float2(v1.x - v3.x, v1.y - v3.y);
  float2 it3 = make_float2(-t3.y, t3.x);   // +i * t3  (inverse transform)

  int base = ((j >> lg) << (lg + 2)) + t;
  dst[base]          = make_float2(t0.x + t2.x, t0.y + t2.y);
  dst[base + ns]     = make_float2(t1.x + it3.x, t1.y + it3.y);
  dst[base + 2 * ns] = make_float2(t0.x - t2.x, t0.y - t2.y);
  dst[base + 3 * ns] = make_float2(t1.x - it3.x, t1.y - it3.y);
}

// WT[n*1024 + k] = Re(F[k*cols + n]) * scale / Npad  (bf16, transposed weight)
__global__ void finalize_wT(const float2* __restrict__ F, const float* __restrict__ scale,
                            u16* __restrict__ WT, int cols, float invn, int total) {
  float sc = scale[0] * invn;
  for (int o = blockIdx.x * blockDim.x + threadIdx.x; o < total;
       o += gridDim.x * blockDim.x) {
    int n = o >> 10;
    int k = o & 1023;
    WT[o] = f2bf(F[(size_t)k * cols + n].x * sc);
  }
}

__global__ void x_to_bf16(const float* __restrict__ x, u16* __restrict__ xb, int n) {
  for (int i = blockIdx.x * blockDim.x + threadIdx.x; i < n;
       i += gridDim.x * blockDim.x) {
    xb[i] = f2bf(x[i]);
  }
}

// ---------------------------------------------------------------------------
// Pipelined 16x64 GEMM core: ping/pong fragment buffers, rolled loop.
// ---------------------------------------------------------------------------

__device__ __forceinline__ void load_btile(v16bf b[4], const u16* const brow[4],
                                           int k, int g) {
#pragma unroll
  for (int nt = 0; nt < 4; ++nt) b[nt] = load_bfrag(brow[nt] + k, g);
}

__device__ __forceinline__ void gemm_16x64(const u16* __restrict__ arow,
                                           const u16* const brow[4], int g,
                                           v8f acc[4]) {
  v16bf a0, a1;
  v16bf b0[4], b1[4];
  a0 = load_afrag(arow, g);
  load_btile(b0, brow, 0, g);

#pragma unroll 1
  for (int k0 = 0; k0 < D_MODEL - 64; k0 += 64) {
    // phase 0: prefetch k0+32, compute on k0
    a1 = load_afrag(arow + k0 + 32, g);
    load_btile(b1, brow, k0 + 32, g);
#pragma unroll
    for (int nt = 0; nt < 4; ++nt) acc[nt] = wmma_bf16(a0, b0[nt], acc[nt]);
    // phase 1: prefetch k0+64, compute on k0+32
    a0 = load_afrag(arow + k0 + 64, g);
    load_btile(b0, brow, k0 + 64, g);
#pragma unroll
    for (int nt = 0; nt < 4; ++nt) acc[nt] = wmma_bf16(a1, b1[nt], acc[nt]);
  }
  // tail: buffers b0/a0 hold k = D_MODEL-64; last k-step is D_MODEL-32
  a1 = load_afrag(arow + D_MODEL - 32, g);
  load_btile(b1, brow, D_MODEL - 32, g);
#pragma unroll
  for (int nt = 0; nt < 4; ++nt) acc[nt] = wmma_bf16(a0, b0[nt], acc[nt]);
#pragma unroll
  for (int nt = 0; nt < 4; ++nt) acc[nt] = wmma_bf16(a1, b1[nt], acc[nt]);
}

// ---------------------------------------------------------------------------
// qkv GEMM: [4096,1024] x [1024,3072] + bias, scatter into Q, K, V^T (bf16)
// ---------------------------------------------------------------------------

__global__ __launch_bounds__(256) void qkv_gemm(
    const u16* __restrict__ xb, const u16* __restrict__ WaT,
    const float* __restrict__ bias, u16* __restrict__ Qb, u16* __restrict__ Kb,
    u16* __restrict__ VTb) {
  int wid  = blockIdx.x * 8 + (threadIdx.x >> 5);   // 12288 waves
  int lane = threadIdx.x & 31;
  int mm = lane & 15, g = lane >> 4;
  int rt = wid / 48, ct = wid % 48;                 // 256 row x 48 col tiles
  int r0 = rt << 4, c0 = ct << 6;

  const u16* arow = xb + (size_t)(r0 + mm) * D_MODEL;
  const u16* brow[4];
#pragma unroll
  for (int nt = 0; nt < 4; ++nt)
    brow[nt] = WaT + (size_t)(c0 + nt * 16 + mm) * D_MODEL;

  v8f acc[4] = {};
  gemm_16x64(arow, brow, g, acc);

#pragma unroll
  for (int nt = 0; nt < 4; ++nt) {
    int col = c0 + nt * 16 + mm;
    float bv = bias[col];
    int sec = col >> 10;
    int cc = col & 1023;
    int h = cc >> 6, d = cc & 63;
#pragma unroll
    for (int i = 0; i < 8; ++i) {
      int row = r0 + i + 8 * g;                     // C layout: M = i + 8g
      int bb = row >> 11, ss = row & 2047;
      int bh = bb * N_HEAD + h;
      u16 val = f2bf(acc[nt][i] + bv);
      if (sec == 0)      Qb[((size_t)bh * SEQ + ss) * HD + d] = val;
      else if (sec == 1) Kb[((size_t)bh * SEQ + ss) * HD + d] = val;
      else               VTb[((size_t)bh * HD + d) * SEQ + ss] = val;
    }
  }
}

// ---------------------------------------------------------------------------
// Flash attention, one wave per (b,h, 16-query tile). Computes S^T = K*Q^T so
// key-softmax is an in-lane reduction + one shfl_xor(16); P^T becomes a B-frag
// after one lane-pair exchange. O^T = V^T * P^T. V loads issue before the
// S-WMMAs; next-tile K fragments prefetch behind them. Loop kept rolled.
// ---------------------------------------------------------------------------

__global__ __launch_bounds__(256) void attention(
    const u16* __restrict__ Qb, const u16* __restrict__ Kb,
    const u16* __restrict__ VTb, u16* __restrict__ AO) {
  int wid  = blockIdx.x * 8 + (threadIdx.x >> 5);  // 4096 waves
  int lane = threadIdx.x & 31;
  int mm = lane & 15, g = lane >> 4;
  int bh = wid >> 7;            // 0..31
  int qt = wid & 127;           // 0..127
  int q0 = qt << 4;

  const u16* Qp = Qb + (size_t)bh * SEQ * HD;
  const u16* Kp = Kb + (size_t)bh * SEQ * HD;
  const u16* Vp = VTb + (size_t)bh * HD * SEQ;

  // Q as B-frags (lane n holds query q0+n), two K-dim steps (d 0-31, 32-63)
  v16bf qf0 = load_bfrag(Qp + (size_t)(q0 + mm) * HD + 0, g);
  v16bf qf1 = load_bfrag(Qp + (size_t)(q0 + mm) * HD + 32, g);

  v8f o0 = {}, o1 = {}, o2 = {}, o3 = {};
  float mrun = -1e30f, lrun = 0.f;

  // preload K fragments for the first key tile
  v16bf kcur[4], knxt[4];
  kcur[0] = load_afrag(Kp + (size_t)(0 + mm) * HD + 0, g);
  kcur[1] = load_afrag(Kp + (size_t)(0 + mm) * HD + 32, g);
  kcur[2] = load_afrag(Kp + (size_t)(16 + mm) * HD + 0, g);
  kcur[3] = load_afrag(Kp + (size_t)(16 + mm) * HD + 32, g);
#pragma unroll
  for (int i = 0; i < 4; ++i) knxt[i] = kcur[i];

#pragma unroll 1
  for (int kt = 0; kt < SEQ / 32; ++kt) {
    int k0 = kt * 32;

    // V fragments for this tile: independent of S, issue first so their
    // latency is covered by the S-WMMAs and the softmax VALU work below.
    v16bf vf0 = load_afrag(Vp + (size_t)(0  + mm) * SEQ + k0, g);
    v16bf vf1 = load_afrag(Vp + (size_t)(16 + mm) * SEQ + k0, g);
    v16bf vf2 = load_afrag(Vp + (size_t)(32 + mm) * SEQ + k0, g);
    v16bf vf3 = load_afrag(Vp + (size_t)(48 + mm) * SEQ + k0, g);

    // S^T tiles: keys 0-15 (s0) and 16-31 (s1) of the tile x 16 queries
    v8f s0 = {}, s1 = {};
    s0 = wmma_bf16(kcur[0], qf0, s0);
    s0 = wmma_bf16(kcur[1], qf1, s0);
    s1 = wmma_bf16(kcur[2], qf0, s1);
    s1 = wmma_bf16(kcur[3], qf1, s1);

    // prefetch next tile's K fragments
    if (kt + 1 < SEQ / 32) {
      int kn = k0 + 32;
      knxt[0] = load_afrag(Kp + (size_t)(kn + mm) * HD + 0, g);
      knxt[1] = load_afrag(Kp + (size_t)(kn + mm) * HD + 32, g);
      knxt[2] = load_afrag(Kp + (size_t)(kn + 16 + mm) * HD + 0, g);
      knxt[3] = load_afrag(Kp + (size_t)(kn + 16 + mm) * HD + 32, g);
    }

    s0 = s0 * 0.125f;   // 1/sqrt(64)
    s1 = s1 * 0.125f;

    float tmax = -1e30f;
#pragma unroll
    for (int i = 0; i < 8; ++i) tmax = fmaxf(tmax, fmaxf(s0[i], s1[i]));
    tmax = fmaxf(tmax, __shfl_xor(tmax, 16, 32));
    float mnew = fmaxf(mrun, tmax);
    float alpha = __expf(mrun - mnew);

    float e0[8], e1[8];
    float lsum = 0.f;
#pragma unroll
    for (int i = 0; i < 8; ++i) {
      e0[i] = __expf(s0[i] - mnew);
      e1[i] = __expf(s1[i] - mnew);
      lsum += e0[i] + e1[i];
    }
    lsum += __shfl_xor(lsum, 16, 32);
    lrun = lrun * alpha + lsum;
    mrun = mnew;
    o0 = o0 * alpha; o1 = o1 * alpha; o2 = o2 * alpha; o3 = o3 * alpha;

    // Build P^T B-frag: lane n+16g holds query n, keys 16g..16g+15.
    Frag pf;
#pragma unroll
    for (int i = 0; i < 8; ++i) {
      float r0v = __shfl_xor(e0[i], 16, 32);   // partner's keys 8+i
      float r1v = __shfl_xor(e1[i], 16, 32);   // partner's keys 16+i
      float lo = g ? r1v : e0[i];
      float hi = g ? e1[i] : r0v;
      pf.u[i]     = f2bf(lo);
      pf.u[8 + i] = f2bf(hi);
    }

    // O^T += V^T(16d x 32k) * P^T(32k x 16q), four d tiles
    o0 = wmma_bf16(vf0, pf.v, o0);
    o1 = wmma_bf16(vf1, pf.v, o1);
    o2 = wmma_bf16(vf2, pf.v, o2);
    o3 = wmma_bf16(vf3, pf.v, o3);

#pragma unroll
    for (int i = 0; i < 4; ++i) kcur[i] = knxt[i];
  }

  float inv = 1.0f / lrun;
  int bb = bh >> 4, h = bh & 15;
  u16* dst = AO + ((size_t)(bb * SEQ + q0 + mm)) * D_MODEL + h * HD;
#pragma unroll
  for (int i = 0; i < 8; ++i) {
    // C layout: VGPR i -> d = dbase + i + 8g, query = mm
    dst[0  + i + 8 * g] = f2bf(o0[i] * inv);
    dst[16 + i + 8 * g] = f2bf(o1[i] * inv);
    dst[32 + i + 8 * g] = f2bf(o2[i] * inv);
    dst[48 + i + 8 * g] = f2bf(o3[i] * inv);
  }
}

// ---------------------------------------------------------------------------
// proj GEMM: [4096,1024] x [1024,1024] + bias -> fp32 output (pipelined)
// ---------------------------------------------------------------------------

__global__ __launch_bounds__(256) void proj_gemm(
    const u16* __restrict__ AO, const u16* __restrict__ WpT,
    const float* __restrict__ bias, float* __restrict__ out) {
  int wid  = blockIdx.x * 8 + (threadIdx.x >> 5);   // 4096 waves
  int lane = threadIdx.x & 31;
  int mm = lane & 15, g = lane >> 4;
  int rt = wid >> 4, ct = wid & 15;                 // 256 row x 16 col tiles
  int r0 = rt << 4, c0 = ct << 6;

  const u16* arow = AO + (size_t)(r0 + mm) * D_MODEL;
  const u16* brow[4];
#pragma unroll
  for (int nt = 0; nt < 4; ++nt)
    brow[nt] = WpT + (size_t)(c0 + nt * 16 + mm) * D_MODEL;

  v8f acc[4] = {};
  gemm_16x64(arow, brow, g, acc);

#pragma unroll
  for (int nt = 0; nt < 4; ++nt) {
    int col = c0 + nt * 16 + mm;
    float bv = bias[col];
#pragma unroll
    for (int i = 0; i < 8; ++i) {
      int row = r0 + i + 8 * g;
      out[(size_t)row * D_MODEL + col] = acc[nt][i] + bv;
    }
  }
}

// ---------------------------------------------------------------------------
// Host launcher
// ---------------------------------------------------------------------------

extern "C" void kernel_launch(void* const* d_in, const int* in_sizes, int n_in,
                              void* d_out, int out_size, void* d_ws, size_t ws_size,
                              hipStream_t stream) {
  const float* x      = (const float*)d_in[0];
  const float* are    = (const float*)d_in[1];
  const float* aim    = (const float*)d_in[2];
  const int*   aidx   = (const int*)d_in[3];
  const float* ascale = (const float*)d_in[4];
  const float* pre    = (const float*)d_in[5];
  const float* pim    = (const float*)d_in[6];
  const int*   pidx   = (const int*)d_in[7];
  const float* pscale = (const float*)d_in[8];
  const float* abias  = (const float*)d_in[9];
  const float* pbias  = (const float*)d_in[10];
  float* out = (float*)d_out;

  char* w = (char*)d_ws;
  float2* fftA = (float2*)(w);                       // 33,554,432 B
  float2* fftB = (float2*)(w + 33554432ull);         // 33,554,432 B
  u16* WaT = (u16*)(w + 67108864ull);                //  6,291,456 B
  u16* WpT = (u16*)(w + 73400320ull);                //  2,097,152 B
  // Activation buffers reuse the FFT ping-pong region (stream-ordered after FFTs):
  u16* xb  = (u16*)(w + 0ull);                       //  8,388,608 B
  u16* Qb  = (u16*)(w + 8388608ull);
  u16* Kb  = (u16*)(w + 16777216ull);
  u16* VTb = (u16*)(w + 25165824ull);
  u16* AO  = (u16*)(w + 33554432ull);                // in fftB region

  // ---- decompress c_attn weight (N = 2^22, 11 radix-4 passes) -> WaT (bf16)
  zero_c<<<ATTN_PAD / 256, 256, 0, stream>>>(fftA, ATTN_PAD);
  scatter_herm<<<ATTN_KEEP / 256, 256, 0, stream>>>(are, aim, aidx, fftA,
                                                    ATTN_KEEP, ATTN_PAD);
  {
    float2 *s = fftA, *d = fftB;
    for (int p = 0; p < 11; ++p) {
      ifft_r4<<<(ATTN_PAD / 4) / 256, 256, 0, stream>>>(s, d, ATTN_PAD, 2 * p);
      float2* t = s; s = d; d = t;
    }
    // 11 passes -> result in fftB (== s)
    finalize_wT<<<2048, 256, 0, stream>>>(s, ascale, WaT, 3 * D_MODEL,
                                          1.0f / (float)ATTN_PAD, ATTN_N);
  }

  // ---- decompress c_proj weight (N = 2^20, 10 radix-4 passes) -> WpT (bf16)
  zero_c<<<PROJ_PAD / 256, 256, 0, stream>>>(fftA, PROJ_PAD);
  scatter_herm<<<PROJ_KEEP / 256, 256, 0, stream>>>(pre, pim, pidx, fftA,
                                                    PROJ_KEEP, PROJ_PAD);
  {
    float2 *s = fftA, *d = fftB;
    for (int p = 0; p < 10; ++p) {
      ifft_r4<<<(PROJ_PAD / 4) / 256, 256, 0, stream>>>(s, d, PROJ_PAD, 2 * p);
      float2* t = s; s = d; d = t;
    }
    // 10 passes -> result in fftA (== s)
    finalize_wT<<<1024, 256, 0, stream>>>(s, pscale, WpT, D_MODEL,
                                          1.0f / (float)PROJ_PAD, PROJ_N);
  }

  // ---- activations
  x_to_bf16<<<4096, 256, 0, stream>>>(x, xb, ROWS * D_MODEL);

  // qkv: 256 row tiles x 48 col tiles = 12288 waves, 8 waves/block
  qkv_gemm<<<1536, 256, 0, stream>>>(xb, WaT, abias, Qb, Kb, VTb);

  // attention: 32 (b,h) x 128 query tiles = 4096 waves
  attention<<<512, 256, 0, stream>>>(Qb, Kb, VTb, AO);

  // proj: 256 x 16 = 4096 waves
  proj_gemm<<<512, 256, 0, stream>>>(AO, WpT, pbias, out);
}
